// GINGRNATTEncoder_81458349736451
// MI455X (gfx1250) — matrix operand order repaired and checked by reading, not compiled, not run on hardware
//
#include <hip/hip_runtime.h>
#include <hip/hip_bf16.h>

// ---------------------------------------------------------------------------
// GIN + GAT + GatedResidual GNN encoder for MI455X (gfx1250, wave32, WMMA).
// GEMMs: v_wmma_f32_16x16x32_bf16, 4 accumulators (16x64 strip) per wave,
//        pre-packed bf16 B fragments (b128 loads), float4 A loads.
// Edge ops: f32 global atomics (scatter-add / softmax max+sum), coalesced.
// ---------------------------------------------------------------------------

typedef __attribute__((ext_vector_type(16))) __bf16 v16bf;
typedef __attribute__((ext_vector_type(8)))  float  v8f;
typedef __attribute__((ext_vector_type(4)))  float  v4f;

#define NNODES 50000
#define NEDGES 600000
#define DIN    64
#define HID    128
#define NHEADS 4
#define HD     32
#define LAT    64
#define MTILES (NNODES / 16)   // 3125

// ---------------------------- small utility kernels ------------------------

__global__ void fill_kernel(float* __restrict__ p, float v, long n) {
  long i = (long)blockIdx.x * blockDim.x + threadIdx.x;
  long s = (long)gridDim.x * blockDim.x;
  for (; i < n; i += s) p[i] = v;
}

// g = (1 + eps) * h   (eps is a device scalar)
__global__ void gin_init_kernel(float* __restrict__ g, const float* __restrict__ h,
                                const float* __restrict__ eps_p, long n) {
  float c = 1.0f + eps_p[0];
  long i = (long)blockIdx.x * blockDim.x + threadIdx.x;
  long s = (long)gridDim.x * blockDim.x;
  for (; i < n; i += s) g[i] = c * h[i];
}

// g[dst[e], :] += h[src[e], :]   (32 lanes per edge, C = 64 or 128 features)
__global__ void scatter_add_kernel(float* __restrict__ g, const float* __restrict__ h,
                                   const int* __restrict__ src, const int* __restrict__ dst,
                                   int C, long total) {
  long i = (long)blockIdx.x * blockDim.x + threadIdx.x;
  long s = (long)gridDim.x * blockDim.x;
  for (; i < total; i += s) {
    long e = i >> 5;
    int lane = (int)(i & 31);
    long sr = src[e], dr = dst[e];
    for (int f = lane; f < C; f += 32)
      atomicAdd(&g[dr * C + f], h[sr * C + f]);
  }
}

// -------------------- weight packing into B-fragment order -----------------
// pk[((nt*(K/32)+ks)*32 + lane)*16 + s] = bf16(W[(ks*32 + (s>>3)*16 + half*8 + (s&7))*NC + nt*16 + l16])
// (B 32x16 bf16, table order: lanes 0-15 hold K 0..7,16..23, lanes 16-31 hold K 8..15,24..31)
__global__ void pack_w_kernel(const float* __restrict__ W, __bf16* __restrict__ pk,
                              int K, int NC, long total) {
  long i = (long)blockIdx.x * blockDim.x + threadIdx.x;
  long st = (long)gridDim.x * blockDim.x;
  int ksteps = K >> 5;
  for (; i < total; i += st) {
    int lane = (int)(i & 31);
    long rest = i >> 5;
    int ks = (int)(rest % ksteps);
    int nt = (int)(rest / ksteps);
    int half = lane >> 4, l16 = lane & 15;
    int col = nt * 16 + l16;
    v16bf b;
#pragma unroll
    for (int v = 0; v < 8; ++v)
#pragma unroll
      for (int t = 0; t < 2; ++t) {
        int k = ks * 32 + ((v >> 2) << 4) + half * 8 + ((2 * v + t) & 7);
        b[2 * v + t] = (__bf16)W[(long)k * NC + col];
      }
    *(v16bf*)(pk + i * 16) = b;
  }
}

// ------------------------------ WMMA GEMM ----------------------------------
// C[M x NC] = act( A[M x K] @ W[K x NC] + bias ),  A split across (A0 | A1) at K0.
// One wave computes a 16x64 strip: one A fragment drives 4 independent WMMAs.
__global__ __launch_bounds__(128)
void wmma_gemm_kernel(const float* __restrict__ A0, const float* __restrict__ A1,
                      int lda0, int lda1, int K0, int K,
                      const __bf16* __restrict__ Wpk, const float* __restrict__ bias,
                      float* __restrict__ C, int NC, int ldC, int act) {
  const int mt = blockIdx.x * 4 + threadIdx.y;      // 16-row tile index
  if (mt >= MTILES) return;                         // wave-uniform: EXEC stays full
  const int m0   = mt * 16;
  const int nt0  = blockIdx.y * 4;                  // first of 4 n-tiles
  const int lane = threadIdx.x;
  const int half = lane >> 4;
  const int l16  = lane & 15;
  const long rowA = m0 + l16;
  const int ksteps = K >> 5;

  v8f acc0 = {}, acc1 = {}, acc2 = {}, acc3 = {};
  for (int ks = 0; ks < ksteps; ++ks) {
    const int k0 = ks * 32;
    // ---- A fragment: 16x32 bf16, K-contiguous float4 loads ----
    const float* Ar;
    int kk;
    if (k0 < K0) { Ar = A0 + rowA * lda0; kk = k0; }
    else         { Ar = A1 + rowA * lda1; kk = k0 - K0; }
    v4f q0 = *(const v4f*)(Ar + kk + half * 8);
    v4f q1 = *(const v4f*)(Ar + kk + half * 8 + 4);
    v4f q2 = *(const v4f*)(Ar + kk + 16 + half * 8);
    v4f q3 = *(const v4f*)(Ar + kk + 16 + half * 8 + 4);
    v16bf a;
#pragma unroll
    for (int j = 0; j < 4; ++j) {
      a[j]      = (__bf16)q0[j];
      a[4 + j]  = (__bf16)q1[j];
      a[8 + j]  = (__bf16)q2[j];
      a[12 + j] = (__bf16)q3[j];
    }
    // ---- 4 B fragments (pre-packed, 32B vector loads) + 4 WMMAs ----
    const __bf16* bp = Wpk + ((long)(nt0 * ksteps + ks) * 32 + lane) * 16;
    const long bstride = (long)ksteps * 32 * 16;
    v16bf b0 = *(const v16bf*)(bp);
    v16bf b1 = *(const v16bf*)(bp + bstride);
    v16bf b2 = *(const v16bf*)(bp + 2 * bstride);
    v16bf b3 = *(const v16bf*)(bp + 3 * bstride);
    acc0 = __builtin_amdgcn_wmma_f32_16x16x32_bf16(false, a, false, b0, (short)0, acc0, false, false);
    acc1 = __builtin_amdgcn_wmma_f32_16x16x32_bf16(false, a, false, b1, (short)0, acc1, false, false);
    acc2 = __builtin_amdgcn_wmma_f32_16x16x32_bf16(false, a, false, b2, (short)0, acc2, false, false);
    acc3 = __builtin_amdgcn_wmma_f32_16x16x32_bf16(false, a, false, b3, (short)0, acc3, false, false);
  }
  // ---- epilogue: D layout = VGPR r -> row r (lanes 0-15) / r+8 (lanes 16-31) ----
  v8f accs[4] = {acc0, acc1, acc2, acc3};
#pragma unroll
  for (int j = 0; j < 4; ++j) {
    int colB = (nt0 + j) * 16 + l16;
    float bv = bias ? bias[colB] : 0.0f;
#pragma unroll
    for (int r = 0; r < 8; ++r) {
      long row = m0 + half * 8 + r;
      float v = accs[j][r] + bv;
      if (act == 1) v = 1.0f / (1.0f + expf(-v));   // sigmoid
      C[row * ldC + colB] = v;
    }
  }
}

// ------------------------- BatchNorm (training stats) ----------------------

__global__ __launch_bounds__(128)
void colstats_kernel(const float* __restrict__ X, float* __restrict__ stats, int rows) {
  int col = threadIdx.x;             // 128 threads = 128 channels
  float s = 0.0f, q = 0.0f;
  for (long r = blockIdx.x; r < rows; r += gridDim.x) {
    float v = X[r * HID + col];
    s += v;
    q += v * v;
  }
  atomicAdd(&stats[col], s);
  atomicAdd(&stats[HID + col], q);
}

__global__ void bn_relu_kernel(float* __restrict__ X, const float* __restrict__ stats,
                               const float* __restrict__ g, const float* __restrict__ b,
                               float inv_rows, long n) {
  long i = (long)blockIdx.x * blockDim.x + threadIdx.x;
  long st = (long)gridDim.x * blockDim.x;
  for (; i < n; i += st) {
    int col = (int)(i & (HID - 1));
    float m  = stats[col] * inv_rows;
    float vr = stats[HID + col] * inv_rows - m * m;
    float v  = (X[i] - m) * rsqrtf(vr + 1e-5f) * g[col] + b[col];
    X[i] = fmaxf(v, 0.0f);
  }
}

// ------------------------------- GAT pieces --------------------------------

__global__ void attn_coef_kernel(const float* __restrict__ xw,
                                 const float* __restrict__ att_src,
                                 const float* __restrict__ att_dst,
                                 float* __restrict__ asrc, float* __restrict__ adst,
                                 long total) {
  long i = (long)blockIdx.x * blockDim.x + threadIdx.x;
  long st = (long)gridDim.x * blockDim.x;
  for (; i < total; i += st) {
    int h = (int)(i & (NHEADS - 1));
    long n = i >> 2;
    const float* row = xw + n * HID + h * HD;
    float s = 0.0f, d = 0.0f;
    for (int j = 0; j < HD; ++j) {
      float v = row[j];
      s += v * att_src[h * HD + j];
      d += v * att_dst[h * HD + j];
    }
    asrc[i] = s;
    adst[i] = d;
  }
}

__device__ __forceinline__ void atomicMaxF(float* addr, float val) {
  if (val >= 0.0f) atomicMax((int*)addr, __float_as_int(val));
  else             atomicMin((unsigned int*)addr, __float_as_uint(val));
}

// pass A: e = leaky_relu(a_src[src] + a_dst[dst], 0.2); emax[dst] = max(...)
__global__ void edge_score_kernel(const float* __restrict__ asrc, const float* __restrict__ adst,
                                  const int* __restrict__ src, const int* __restrict__ dst,
                                  float* __restrict__ ebuf, float* __restrict__ emax, long total) {
  long i = (long)blockIdx.x * blockDim.x + threadIdx.x;
  long st = (long)gridDim.x * blockDim.x;
  for (; i < total; i += st) {
    long e = i >> 2;
    int  h = (int)(i & (NHEADS - 1));
    long sr = src[e], dr = dst[e];
    float v = asrc[sr * NHEADS + h] + adst[dr * NHEADS + h];
    v = (v > 0.0f) ? v : 0.2f * v;
    ebuf[i] = v;
    atomicMaxF(&emax[dr * NHEADS + h], v);
  }
}

// pass B: ex = exp(e - emax[dst]); denom[dst] += ex
__global__ void edge_exp_kernel(float* __restrict__ ebuf, const float* __restrict__ emax,
                                const int* __restrict__ dst, float* __restrict__ denom, long total) {
  long i = (long)blockIdx.x * blockDim.x + threadIdx.x;
  long st = (long)gridDim.x * blockDim.x;
  for (; i < total; i += st) {
    long e = i >> 2;
    int  h = (int)(i & (NHEADS - 1));
    long dr = dst[e];
    float v = expf(ebuf[i] - emax[dr * NHEADS + h]);
    ebuf[i] = v;
    atomicAdd(&denom[dr * NHEADS + h], v);
  }
}

// pass C: attn[dst, :] += alpha(head) * xw[src, :]   (32 lanes/edge, 4 floats/lane)
__global__ void edge_agg_kernel(const float* __restrict__ ebuf, const float* __restrict__ denom,
                                const int* __restrict__ src, const int* __restrict__ dst,
                                const float* __restrict__ xw, float* __restrict__ attn, long total) {
  long i = (long)blockIdx.x * blockDim.x + threadIdx.x;
  long st = (long)gridDim.x * blockDim.x;
  for (; i < total; i += st) {
    long e = i >> 5;
    int  q = (int)(i & 31);
    int  h = q >> 3;                    // head
    int  f0 = q * 4;                    // = h*HD + (q&7)*4 since HD=32
    long sr = src[e], dr = dst[e];
    float alpha = ebuf[e * NHEADS + h] / (denom[dr * NHEADS + h] + 1e-16f);
    const float* xr = xw + sr * HID + f0;
    float* ar = attn + dr * HID + f0;
#pragma unroll
    for (int j = 0; j < 4; ++j)
      atomicAdd(&ar[j], alpha * xr[j]);
  }
}

// ----------------------------- LayerNorm (per row) -------------------------
__global__ __launch_bounds__(128)
void layernorm_kernel(float* __restrict__ out, const float* __restrict__ a,
                      const float* __restrict__ badd, const float* __restrict__ biasvec,
                      const float* __restrict__ g, const float* __restrict__ b,
                      int relu, int nrows) {
  int row = blockIdx.x * blockDim.y + threadIdx.y;
  if (row >= nrows) return;
  int lane = threadIdx.x;
  float vals[4];
  float s = 0.0f;
#pragma unroll
  for (int j = 0; j < 4; ++j) {
    int c = lane + 32 * j;
    float v = a[(long)row * HID + c];
    if (badd)    v += badd[(long)row * HID + c];
    if (biasvec) v += biasvec[c];
    vals[j] = v;
    s += v;
  }
#pragma unroll
  for (int off = 16; off; off >>= 1) s += __shfl_xor(s, off, 32);
  float mean = s * (1.0f / HID);
  float q = 0.0f;
#pragma unroll
  for (int j = 0; j < 4; ++j) { float d = vals[j] - mean; q += d * d; }
#pragma unroll
  for (int off = 16; off; off >>= 1) q += __shfl_xor(q, off, 32);
  float inv = rsqrtf(q * (1.0f / HID) + 1e-5f);
#pragma unroll
  for (int j = 0; j < 4; ++j) {
    int c = lane + 32 * j;
    float v = (vals[j] - mean) * inv * g[c] + b[c];
    if (relu) v = fmaxf(v, 0.0f);
    out[(long)row * HID + c] = v;
  }
}

__global__ void gated_kernel(float* __restrict__ h, const float* __restrict__ gate,
                             const float* __restrict__ t, const float* __restrict__ hprev, long n) {
  long i = (long)blockIdx.x * blockDim.x + threadIdx.x;
  long st = (long)gridDim.x * blockDim.x;
  for (; i < n; i += st) {
    float gv = gate[i];
    h[i] = gv * t[i] + (1.0f - gv) * hprev[i];
  }
}

__global__ void final_kernel(const float* __restrict__ mean, float* __restrict__ logstd,
                             float* __restrict__ z, const float* __restrict__ noise, long n) {
  long i = (long)blockIdx.x * blockDim.x + threadIdx.x;
  long st = (long)gridDim.x * blockDim.x;
  for (; i < n; i += st) {
    float ls = tanhf(logstd[i]) * 5.0f;
    logstd[i] = ls;
    z[i] = mean[i] + noise[i] * expf(ls);
  }
}

// --------------------------------- host ------------------------------------

static void launch_gemm(const float* A0, const float* A1, int lda0, int lda1,
                        int K0, int K, const float* W, __bf16* wpk, const float* bias,
                        float* C, int NC, int ldC, int act, hipStream_t s) {
  long ptotal = (long)(NC / 16) * (K / 32) * 32;
  pack_w_kernel<<<(int)((ptotal + 255) / 256), 256, 0, s>>>(W, wpk, K, NC, ptotal);
  dim3 blk(32, 4);                               // 4 waves/block, 16x64 strip each
  dim3 grd((MTILES + 3) / 4, NC / 64);           // 782 x {2 (NC=128) | 1 (NC=64)}
  wmma_gemm_kernel<<<grd, blk, 0, s>>>(A0, A1, lda0, lda1, K0, K, wpk, bias, C, NC, ldC, act);
}

extern "C" void kernel_launch(void* const* d_in, const int* in_sizes, int n_in,
                              void* d_out, int out_size, void* d_ws, size_t ws_size,
                              hipStream_t stream) {
  (void)in_sizes; (void)n_in; (void)out_size; (void)ws_size;

  const float* x     = (const float*)d_in[0];
  const int*   ei    = (const int*)d_in[1];
  const float* noise = (const float*)d_in[2];
  const int* esrc = ei;
  const int* edst = ei + NEDGES;

  // params: 23 leaves per layer (insertion order), then mean.W/b, logstd.W/b
  const int base = 3, PL = 23;
  auto P = [&](int layer, int idx) -> const float* {
    return (const float*)d_in[base + layer * PL + idx];
  };
  const float* meanW = (const float*)d_in[base + 3 * PL + 0];
  const float* meanB = (const float*)d_in[base + 3 * PL + 1];
  const float* lstdW = (const float*)d_in[base + 3 * PL + 2];
  const float* lstdB = (const float*)d_in[base + 3 * PL + 3];

  // workspace layout (floats)
  const long NH = (long)NNODES * HID;
  float* ws    = (float*)d_ws;
  float* hcur  = ws;                       // N x 128 current features
  float* B0    = hcur + NH;                // g, later gate
  float* B1    = B0 + NH;                  // lin1 out / grn_t1 out
  float* B2    = B1 + NH;                  // h_prev
  float* B3    = B2 + NH;                  // xw, later t
  float* B4    = B3 + NH;                  // attn agg -> h_attn
  float* ebuf  = B4 + NH;                  // E x 4
  float* asrc  = ebuf + (long)NEDGES * NHEADS;
  float* adst  = asrc + (long)NNODES * NHEADS;
  float* emax  = adst + (long)NNODES * NHEADS;
  float* denom = emax + (long)NNODES * NHEADS;
  float* stats = denom + (long)NNODES * NHEADS;   // 256 floats (sum | sumsq)
  __bf16* wpk  = (__bf16*)(stats + 2 * HID);      // packed weights, <= 64 KB

  const long nNH = (long)NNODES * NHEADS;
  const long eNH = (long)NEDGES * NHEADS;
  const float invN = 1.0f / (float)NNODES;

  for (int l = 0; l < 3; ++l) {
    const float* hin = (l == 0) ? x : hcur;
    const int Cin = (l == 0) ? DIN : HID;
    const long nC = (long)NNODES * Cin;

    // ---- GIN: g = (1+eps)*h, then scatter-add of neighbors ----
    gin_init_kernel<<<2048, 256, 0, stream>>>(B0, hin, P(l, 0), nC);
    scatter_add_kernel<<<8192, 256, 0, stream>>>(B0, hin, esrc, edst, Cin, (long)NEDGES * 32);

    // lin1 -> BN -> relu
    launch_gemm(B0, B0, Cin, Cin, Cin, Cin, P(l, 1), wpk, P(l, 2), B1, HID, HID, 0, stream);
    fill_kernel<<<1, 256, 0, stream>>>(stats, 0.0f, 2 * HID);
    colstats_kernel<<<512, HID, 0, stream>>>(B1, stats, NNODES);
    bn_relu_kernel<<<2048, 256, 0, stream>>>(B1, stats, P(l, 3), P(l, 4), invN, NH);

    // lin2 -> BN -> relu  => h_prev in B2
    launch_gemm(B1, B1, HID, HID, HID, HID, P(l, 5), wpk, P(l, 6), B2, HID, HID, 0, stream);
    fill_kernel<<<1, 256, 0, stream>>>(stats, 0.0f, 2 * HID);
    colstats_kernel<<<512, HID, 0, stream>>>(B2, stats, NNODES);
    bn_relu_kernel<<<2048, 256, 0, stream>>>(B2, stats, P(l, 7), P(l, 8), invN, NH);

    // ---- GAT ----
    launch_gemm(B2, B2, HID, HID, HID, HID, P(l, 9), wpk, nullptr, B3, HID, HID, 0, stream);
    attn_coef_kernel<<<784, 256, 0, stream>>>(B3, P(l, 10), P(l, 11), asrc, adst, nNH);
    fill_kernel<<<256, 256, 0, stream>>>(emax, -__builtin_huge_valf(), nNH);
    fill_kernel<<<256, 256, 0, stream>>>(denom, 0.0f, nNH);
    edge_score_kernel<<<4096, 256, 0, stream>>>(asrc, adst, esrc, edst, ebuf, emax, eNH);
    edge_exp_kernel<<<4096, 256, 0, stream>>>(ebuf, emax, edst, denom, eNH);
    fill_kernel<<<2048, 256, 0, stream>>>(B4, 0.0f, NH);
    edge_agg_kernel<<<8192, 256, 0, stream>>>(ebuf, denom, esrc, edst, B3, B4, (long)NEDGES * 32);
    {
      dim3 blk(32, 4);
      layernorm_kernel<<<(NNODES + 3) / 4, blk, 0, stream>>>(
          B4, B4, B2, P(l, 12), P(l, 13), P(l, 14), 0, NNODES);
    }

    // ---- GatedResidual ----
    launch_gemm(B4, B2, HID, HID, HID, 2 * HID, P(l, 15), wpk, P(l, 16), B0, HID, HID, 1, stream);
    launch_gemm(B4, B4, HID, HID, HID, HID, P(l, 17), wpk, P(l, 18), B1, HID, HID, 0, stream);
    {
      dim3 blk(32, 4);
      layernorm_kernel<<<(NNODES + 3) / 4, blk, 0, stream>>>(
          B1, B1, nullptr, nullptr, P(l, 19), P(l, 20), 1, NNODES);
    }
    launch_gemm(B1, B1, HID, HID, HID, HID, P(l, 21), wpk, P(l, 22), B3, HID, HID, 0, stream);
    gated_kernel<<<2048, 256, 0, stream>>>(hcur, B0, B3, B2, NH);
  }

  // ---- heads: mean, logstd, z ----
  float* out_mean = (float*)d_out;
  float* out_lstd = out_mean + (long)NNODES * LAT;
  float* out_z    = out_lstd + (long)NNODES * LAT;
  launch_gemm(hcur, hcur, HID, HID, HID, HID, meanW, wpk, meanB, out_mean, LAT, LAT, 0, stream);
  launch_gemm(hcur, hcur, HID, HID, HID, HID, lstdW, wpk, lstdB, out_lstd, LAT, LAT, 0, stream);
  final_kernel<<<2048, 256, 0, stream>>>(out_mean, out_lstd, out_z, noise, (long)NNODES * LAT);
}
